// BoundaryConsistencyModule_67594195305232
// MI455X (gfx1250) — matrix operand order repaired and checked by reading, not compiled, not run on hardware
//
#include <hip/hip_runtime.h>
#include <cstddef>
#include <cstdint>

typedef __attribute__((ext_vector_type(16))) _Float16 v16h;
typedef __attribute__((ext_vector_type(8)))  _Float16 v8h;
typedef __attribute__((ext_vector_type(8)))  float    v8f;

#define N_PTS  192000
#define P_PATCH 4000

static __device__ __forceinline__ v16h cat8(v8h lo, v8h hi) {
    return __builtin_shufflevector(lo, hi, 0,1,2,3,4,5,6,7,8,9,10,11,12,13,14,15);
}
static __device__ __forceinline__ v8h h8zero() {
    v8h z;
#pragma unroll
    for (int i = 0; i < 8; ++i) z[i] = (_Float16)0.0f;
    return z;
}
static __device__ __forceinline__ v16h cat8z(v8h lo) { return cat8(lo, h8zero()); }
static __device__ __forceinline__ v8f f8zero() {
    v8f z;
#pragma unroll
    for (int i = 0; i < 8; ++i) z[i] = 0.0f;
    return z;
}
static __device__ __forceinline__ v8f wmma_f16(v16h a, v16h b, v8f c) {
    return __builtin_amdgcn_wmma_f32_16x16x32_f16(false, a, false, b, (short)0, c, false, false);
}
// A fragment (16x32 f16) from an LDS tile row of 64 halves. hb = (lane>=16)?8:0
static __device__ __forceinline__ v16h a_frag(const _Float16* row, int k0, int hb) {
    const v8h* lo = (const v8h*)(row + k0 + 2 * hb);
    const v8h* hi = (const v8h*)(row + k0 + 2 * hb + 8);
    return cat8(*lo, *hi);
}
// B fragment (32x16 f16) from transposed weights [n][ldk]. koff = ks*32 + ((lane>=16)?16:0)
static __device__ __forceinline__ v16h b_frag(const _Float16* matT, int ldk, int n, int koff) {
    const v8h* p = (const v8h*)(matT + (size_t)n * ldk + koff);
    return cat8(p[0], p[1]);
}
static __device__ __forceinline__ float gelu_exact(float x) {
    return 0.5f * x * (1.0f + erff(x * 0.70710678118654752f));
}
// ---- CDNA5 async global->LDS copy (ASYNCcnt path) ----
static __device__ __forceinline__ unsigned lds_addr32(const void* p) {
    return (unsigned)(unsigned long long)(__attribute__((address_space(3))) const void*)p;
}
static __device__ __forceinline__ void async_copy_b128(unsigned lds_byte_off, const void* gptr) {
    asm volatile("global_load_async_to_lds_b128 %0, %1, off"
                 :: "v"(lds_byte_off), "v"(gptr) : "memory");
}
static __device__ __forceinline__ void wait_async0() {
    asm volatile("s_wait_asynccnt 0" ::: "memory");
}

// ---------------- K0: weight conversion fp32 -> f16 transposed [n][k] ----------------
__global__ __launch_bounds__(256) void k_wprep(const float* __restrict__ Wqkv,
                                               const float* __restrict__ Wattn,
                                               const float* __restrict__ Wf1,
                                               const float* __restrict__ Wf2,
                                               _Float16* __restrict__ wbuf) {
    int idx = blockIdx.x * 256 + threadIdx.x;            // 49152 total
    if (idx < 12288)      { int l = idx;          int n = l / 64,  k = l % 64;  wbuf[idx] = (_Float16)Wqkv[k * 192 + n]; }
    else if (idx < 16384) { int l = idx - 12288;  int n = l / 64,  k = l % 64;  wbuf[idx] = (_Float16)Wattn[k * 64 + n]; }
    else if (idx < 32768) { int l = idx - 16384;  int n = l / 64,  k = l % 64;  wbuf[idx] = (_Float16)Wf1[k * 256 + n]; }
    else if (idx < 49152) { int l = idx - 32768;  int n = l / 256, k = l % 256; wbuf[idx] = (_Float16)Wf2[k * 64 + n]; }
}

// ---------------- K1: softmax(seg)+coord -> feat = X @ W_in + b_in ----------------
__global__ __launch_bounds__(256) void k_inproj(const float* __restrict__ seg,
                                                const float* __restrict__ coord,
                                                const float* __restrict__ Win,
                                                const float* __restrict__ bin,
                                                float* __restrict__ feat) {
    int j = blockIdx.x * 256 + threadIdx.x;
    float x[23];
    float mx = -3.0e38f;
#pragma unroll
    for (int k = 0; k < 20; ++k) { x[k] = seg[(size_t)j * 20 + k]; mx = fmaxf(mx, x[k]); }
    float s = 0.0f;
#pragma unroll
    for (int k = 0; k < 20; ++k) { x[k] = expf(x[k] - mx); s += x[k]; }
    float inv = 1.0f / s;
#pragma unroll
    for (int k = 0; k < 20; ++k) x[k] *= inv;
    x[20] = coord[(size_t)j * 3 + 0];
    x[21] = coord[(size_t)j * 3 + 1];
    x[22] = coord[(size_t)j * 3 + 2];
    for (int c = 0; c < 64; ++c) {
        float acc = bin[c];
#pragma unroll
        for (int k = 0; k < 23; ++k) acc += x[k] * Win[k * 64 + c];
        feat[(size_t)j * 64 + c] = acc;
    }
}

// ---------------- K2: qkv = LN(feat)@Wqkv + b, gathered by order, split to q/k/vT ----------------
__global__ __launch_bounds__(256) void k_qkv(const float* __restrict__ feat,
                                             const int* __restrict__ order,
                                             const float* __restrict__ g1,
                                             const float* __restrict__ be1,
                                             const _Float16* __restrict__ qkvT,   // [192][64]
                                             const float* __restrict__ bqkv,
                                             _Float16* __restrict__ qbuf,         // [P][4][48][16]
                                             _Float16* __restrict__ kbuf,         // [P][4][48][16]
                                             _Float16* __restrict__ vbuf) {       // [P][4][16][48]
    __shared__ __align__(16) _Float16 lds_a[8][16][64];
    const int wave = threadIdx.x >> 5, lane = threadIdx.x & 31;
    const int rowbase = blockIdx.x * 128 + wave * 16;
    if (lane < 16) {
        int src = order[rowbase + lane];
        const float4* f4 = (const float4*)(feat + (size_t)src * 64);
        float4 x[16];
        float s = 0.0f;
#pragma unroll
        for (int t = 0; t < 16; ++t) { x[t] = f4[t]; s += x[t].x + x[t].y + x[t].z + x[t].w; }
        float mean = s * (1.0f / 64.0f), v = 0.0f;
#pragma unroll
        for (int t = 0; t < 16; ++t) {
            float a = x[t].x - mean, b = x[t].y - mean, c = x[t].z - mean, d = x[t].w - mean;
            v += a * a + b * b + c * c + d * d;
        }
        float rstd = rsqrtf(v * (1.0f / 64.0f) + 1e-5f);
#pragma unroll
        for (int t = 0; t < 16; ++t) {
            int c0 = t * 4;
            lds_a[wave][lane][c0 + 0] = (_Float16)((x[t].x - mean) * rstd * g1[c0 + 0] + be1[c0 + 0]);
            lds_a[wave][lane][c0 + 1] = (_Float16)((x[t].y - mean) * rstd * g1[c0 + 1] + be1[c0 + 1]);
            lds_a[wave][lane][c0 + 2] = (_Float16)((x[t].z - mean) * rstd * g1[c0 + 2] + be1[c0 + 2]);
            lds_a[wave][lane][c0 + 3] = (_Float16)((x[t].w - mean) * rstd * g1[c0 + 3] + be1[c0 + 3]);
        }
    }
    __syncthreads();
    const int hb = (lane >= 16) ? 8 : 0;
    const int nloc = lane & 15;
    const int kb = (lane >= 16) ? 16 : 0;
    // tile lies within one patch: 48 = 3*16 and rowbase % 16 == 0
    const int pblk = rowbase / 48;          // wave-uniform (scalar)
    const int sb   = rowbase % 48;          // wave-uniform, in {0,16,32}
    const int moff = (lane >= 16) ? 8 : 0;
    const _Float16* arow = &lds_a[wave][lane & 15][0];
    v16h a0 = a_frag(arow, 0, hb), a1 = a_frag(arow, 32, hb);
#pragma unroll
    for (int ct = 0; ct < 12; ++ct) {
        int ncol = ct * 16 + nloc;
        v8f acc = f8zero();
        acc = wmma_f16(a0, b_frag(qkvT, 64, ncol, 0 + kb), acc);
        acc = wmma_f16(a1, b_frag(qkvT, 64, ncol, 32 + kb), acc);
        float bias = bqkv[ncol];
        int which = ct >> 2, h = ct & 3, d = nloc;
        _Float16* qk_base = (which == 0 ? qbuf : kbuf) + (((size_t)pblk * 4 + h) * 48 + sb + moff) * 16 + d;
        _Float16* v_base  = vbuf + (((size_t)pblk * 4 + h) * 16 + d) * 48 + sb + moff;
#pragma unroll
        for (int vg = 0; vg < 8; ++vg) {
            float val = acc[vg] + bias;
            if (which == 0)      qk_base[(size_t)vg * 16] = (_Float16)(val * 0.25f); // * D^-0.5
            else if (which == 1) qk_base[(size_t)vg * 16] = (_Float16)val;
            else                 v_base[vg]               = (_Float16)val;
        }
    }
}

// ---------------- K3: attention per patch (block=128, wave=head) ----------------
__global__ __launch_bounds__(128) void k_attn(const int* __restrict__ order,
                                              const int* __restrict__ gcoord,
                                              const float* __restrict__ rpe,      // [69][4]
                                              const _Float16* __restrict__ qbuf,
                                              const _Float16* __restrict__ kbuf,
                                              const _Float16* __restrict__ vbuf,
                                              _Float16* __restrict__ obuf) {      // [N][64] serialized
    __shared__ int gc_s[48][3];
    __shared__ float rpe_s[276];
    __shared__ __align__(16) _Float16 probs[4][48][48];
    const int p = blockIdx.x, tid = threadIdx.x;
    for (int t = tid; t < 144; t += 128) {
        int pt = t / 3, dim = t % 3;
        gc_s[pt][dim] = gcoord[(size_t)order[p * 48 + pt] * 3 + dim];
    }
    for (int t = tid; t < 276; t += 128) rpe_s[t] = rpe[t];
    __syncthreads();
    const int wave = tid >> 5, lane = tid & 31, h = wave;
    const int base = (lane >= 16) ? 8 : 0, mloc = lane & 15;
    const _Float16* qh = qbuf + ((size_t)p * 4 + h) * 48 * 16;
    const _Float16* kh = kbuf + ((size_t)p * 4 + h) * 48 * 16;
    const _Float16* vh = vbuf + ((size_t)p * 4 + h) * 16 * 48;
    // Q fragments (K padded 16->32 with zeros)
    v16h aq[3];
#pragma unroll
    for (int mt = 0; mt < 3; ++mt)
        aq[mt] = cat8z(*(const v8h*)(qh + (16 * mt + mloc) * 16 + base));
    // K^T fragments: elements 8..15 of every lane cover padded K-range -> zero
    v16h bk[3];
#pragma unroll
    for (int nt = 0; nt < 3; ++nt) {
        bk[nt] = cat8z(*(const v8h*)(kh + (16 * nt + mloc) * 16 + base));
    }
    // V fragments (mt-invariant): B[k=key][n=d] from vT layout
    v16h vb0, vb1;
    {
        int d = mloc;
        const v8h* pv = (const v8h*)(vh + d * 48 + ((lane >= 16) ? 16 : 0));
        vb0 = cat8(pv[0], pv[1]);
        if (lane < 16) {
            const v8h* pv1 = (const v8h*)(vh + d * 48 + 32);
            vb1 = cat8(pv1[0], pv1[1]);
        } else vb1 = cat8(h8zero(), h8zero());
    }
    v8f s[3][3];
#pragma unroll
    for (int mt = 0; mt < 3; ++mt)
#pragma unroll
        for (int nt = 0; nt < 3; ++nt) {
            v8f acc = wmma_f16(aq[mt], bk[nt], f8zero());
#pragma unroll
            for (int vg = 0; vg < 8; ++vg) {   // + relative position bias
                int m = 16 * mt + vg + ((lane >= 16) ? 8 : 0);
                int n = 16 * nt + mloc;
                float b = 0.0f;
#pragma unroll
                for (int dim = 0; dim < 3; ++dim) {
                    int rel = gc_s[m][dim] - gc_s[n][dim];
                    rel = min(11, max(-11, rel)) + 11 + dim * 23;
                    b += rpe_s[rel * 4 + h];
                }
                acc[vg] += b;
            }
            s[mt][nt] = acc;
        }
    // softmax over 48 keys: rows live in one half-wave (16 lanes) x 3 tiles
#pragma unroll
    for (int mt = 0; mt < 3; ++mt)
#pragma unroll
        for (int vg = 0; vg < 8; ++vg) {
            float v0 = s[mt][0][vg], v1 = s[mt][1][vg], v2 = s[mt][2][vg];
            float mx = fmaxf(v0, fmaxf(v1, v2));
#pragma unroll
            for (int msk = 1; msk < 16; msk <<= 1) mx = fmaxf(mx, __shfl_xor(mx, msk, 32));
            float e0 = expf(v0 - mx), e1 = expf(v1 - mx), e2 = expf(v2 - mx);
            float sm = e0 + e1 + e2;
#pragma unroll
            for (int msk = 1; msk < 16; msk <<= 1) sm += __shfl_xor(sm, msk, 32);
            float inv = 1.0f / sm;
            int m = 16 * mt + vg + ((lane >= 16) ? 8 : 0);
            probs[h][m][0  + mloc] = (_Float16)(e0 * inv);
            probs[h][m][16 + mloc] = (_Float16)(e1 * inv);
            probs[h][m][32 + mloc] = (_Float16)(e2 * inv);
        }
    __syncthreads();
    // out = probs(48x48) @ v(48x16)
#pragma unroll
    for (int mt = 0; mt < 3; ++mt) {
        const _Float16* prow = &probs[h][16 * mt + mloc][0];
        v16h pa0 = cat8(*(const v8h*)(prow + 2 * base), *(const v8h*)(prow + 2 * base + 8));
        v16h pa1 = (lane < 16) ? cat8(*(const v8h*)(prow + 32), *(const v8h*)(prow + 40)) : cat8(h8zero(), h8zero());
        v8f o = wmma_f16(pa0, vb0, f8zero());
        o = wmma_f16(pa1, vb1, o);
#pragma unroll
        for (int vg = 0; vg < 8; ++vg) {
            int m = 16 * mt + vg + ((lane >= 16) ? 8 : 0);
            obuf[((size_t)p * 48 + m) * 64 + h * 16 + mloc] = (_Float16)o[vg];
        }
    }
}

// ---------------- K4a: attn-proj (gather by inverse) + bias + residual -> feat2 ----------------
__global__ __launch_bounds__(256) void k_attnproj(const _Float16* __restrict__ obuf,
                                                  const int* __restrict__ inverse,
                                                  const float* __restrict__ feat,
                                                  const _Float16* __restrict__ attnT, // [64][64]
                                                  const float* __restrict__ battn,
                                                  float* __restrict__ feat2) {
    __shared__ __align__(16) _Float16 lds_a[8][16][64];
    const int wave = threadIdx.x >> 5, lane = threadIdx.x & 31;
    const int rowbase = blockIdx.x * 128 + wave * 16;
    // async global->LDS staging of the gathered 16x64 f16 tile (2KB), all 32 lanes
    {
        unsigned lbase = lds_addr32(&lds_a[wave][0][0]);
#pragma unroll
        for (int c = 0; c < 4; ++c) {
            int q = c * 32 + lane;                 // 16B chunk id, 0..127
            int row = q >> 3, col = q & 7;
            const _Float16* g = obuf + (size_t)inverse[rowbase + row] * 64 + col * 8;
            async_copy_b128(lbase + q * 16, g);
        }
        wait_async0();
    }
    __syncthreads();
    const int hb = (lane >= 16) ? 8 : 0, nloc = lane & 15, kb = (lane >= 16) ? 16 : 0;
    const _Float16* arow = &lds_a[wave][lane & 15][0];
    v16h a0 = a_frag(arow, 0, hb), a1 = a_frag(arow, 32, hb);
#pragma unroll
    for (int ct = 0; ct < 4; ++ct) {
        int n = ct * 16 + nloc;
        v8f acc = f8zero();
        acc = wmma_f16(a0, b_frag(attnT, 64, n, 0 + kb), acc);
        acc = wmma_f16(a1, b_frag(attnT, 64, n, 32 + kb), acc);
        float bias = battn[n];
#pragma unroll
        for (int vg = 0; vg < 8; ++vg) {
            int m = vg + ((lane >= 16) ? 8 : 0);
            size_t j = (size_t)(rowbase + m);
            feat2[j * 64 + n] = acc[vg] + bias + feat[j * 64 + n];
        }
    }
}

// ---------------- K4b: LN2(feat2) -> hbuf f16 ----------------
__global__ __launch_bounds__(256) void k_ln2(const float* __restrict__ feat2,
                                             const float* __restrict__ g2,
                                             const float* __restrict__ be2,
                                             _Float16* __restrict__ hbuf) {
    int j = blockIdx.x * 256 + threadIdx.x;
    const float4* f4 = (const float4*)(feat2 + (size_t)j * 64);
    float4 x[16];
    float s = 0.0f;
#pragma unroll
    for (int t = 0; t < 16; ++t) { x[t] = f4[t]; s += x[t].x + x[t].y + x[t].z + x[t].w; }
    float mean = s * (1.0f / 64.0f), v = 0.0f;
#pragma unroll
    for (int t = 0; t < 16; ++t) {
        float a = x[t].x - mean, b = x[t].y - mean, c = x[t].z - mean, d = x[t].w - mean;
        v += a * a + b * b + c * c + d * d;
    }
    float rstd = rsqrtf(v * (1.0f / 64.0f) + 1e-5f);
    _Float16* hr = hbuf + (size_t)j * 64;
#pragma unroll
    for (int t = 0; t < 16; ++t) {
        int c0 = t * 4;
        hr[c0 + 0] = (_Float16)((x[t].x - mean) * rstd * g2[c0 + 0] + be2[c0 + 0]);
        hr[c0 + 1] = (_Float16)((x[t].y - mean) * rstd * g2[c0 + 1] + be2[c0 + 1]);
        hr[c0 + 2] = (_Float16)((x[t].z - mean) * rstd * g2[c0 + 2] + be2[c0 + 2]);
        hr[c0 + 3] = (_Float16)((x[t].w - mean) * rstd * g2[c0 + 3] + be2[c0 + 3]);
    }
}

// ---------------- K4c: FFN fused: gelu(h@W_f1+b)@W_f2 + b + feat2 -> featF ----------------
__global__ __launch_bounds__(128) void k_ffn(const _Float16* __restrict__ hbuf,
                                             const _Float16* __restrict__ f1T,   // [256][64]
                                             const float* __restrict__ bf1,
                                             const _Float16* __restrict__ f2T,   // [64][256]
                                             const float* __restrict__ bf2,
                                             const float* __restrict__ feat2,
                                             float* __restrict__ featF) {
    __shared__ __align__(16) _Float16 ha[4][16][64];
    __shared__ __align__(16) _Float16 mid[4][16][256];
    const int wave = threadIdx.x >> 5, lane = threadIdx.x & 31;
    const int rowbase = blockIdx.x * 64 + wave * 16;
    // async global->LDS staging of the 16x64 f16 tile (2KB), all 32 lanes
    {
        unsigned lbase = lds_addr32(&ha[wave][0][0]);
#pragma unroll
        for (int c = 0; c < 4; ++c) {
            int q = c * 32 + lane;                 // 16B chunk id, 0..127
            int row = q >> 3, col = q & 7;
            const _Float16* g = hbuf + (size_t)(rowbase + row) * 64 + col * 8;
            async_copy_b128(lbase + q * 16, g);
        }
        wait_async0();
    }
    __syncthreads();
    const int hb = (lane >= 16) ? 8 : 0, nloc = lane & 15, kb = (lane >= 16) ? 16 : 0;
    const _Float16* arow = &ha[wave][lane & 15][0];
    v16h a0 = a_frag(arow, 0, hb), a1 = a_frag(arow, 32, hb);
    // stage 1: 64 -> 256, gelu, into LDS
#pragma unroll
    for (int ct = 0; ct < 16; ++ct) {
        int n = ct * 16 + nloc;
        v8f acc = f8zero();
        acc = wmma_f16(a0, b_frag(f1T, 64, n, 0 + kb), acc);
        acc = wmma_f16(a1, b_frag(f1T, 64, n, 32 + kb), acc);
        float bias = bf1[n];
#pragma unroll
        for (int vg = 0; vg < 8; ++vg) {
            int m = vg + ((lane >= 16) ? 8 : 0);
            mid[wave][m][n] = (_Float16)gelu_exact(acc[vg] + bias);
        }
    }
    __syncthreads();
    // stage 2: 256 -> 64, + bias + residual
    v16h a2[8];
    const _Float16* mrow = &mid[wave][lane & 15][0];
#pragma unroll
    for (int ks = 0; ks < 8; ++ks)
        a2[ks] = cat8(*(const v8h*)(mrow + ks * 32 + 2 * hb), *(const v8h*)(mrow + ks * 32 + 2 * hb + 8));
#pragma unroll
    for (int ct = 0; ct < 4; ++ct) {
        int n = ct * 16 + nloc;
        v8f acc = f8zero();
#pragma unroll
        for (int ks = 0; ks < 8; ++ks)
            acc = wmma_f16(a2[ks], b_frag(f2T, 256, n, ks * 32 + kb), acc);
        float bias = bf2[n];
#pragma unroll
        for (int vg = 0; vg < 8; ++vg) {
            int m = vg + ((lane >= 16) ? 8 : 0);
            size_t j = (size_t)(rowbase + m);
            featF[j * 64 + n] = acc[vg] + bias + feat2[j * 64 + n];
        }
    }
}

// ---------------- K4d: head: gelu(featF@W_h1+b)@W_h2 + b -> out (N,3) ----------------
__global__ __launch_bounds__(256) void k_head(const float* __restrict__ featF,
                                              const float* __restrict__ W1,
                                              const float* __restrict__ b1,
                                              const float* __restrict__ W2,
                                              const float* __restrict__ b2,
                                              float* __restrict__ out) {
    int j = blockIdx.x * 256 + threadIdx.x;
    const float4* f4 = (const float4*)(featF + (size_t)j * 64);
    float4 x[16];
#pragma unroll
    for (int t = 0; t < 16; ++t) x[t] = f4[t];
    float o0 = b2[0], o1 = b2[1], o2 = b2[2];
    for (int n = 0; n < 32; ++n) {
        float s = b1[n];
#pragma unroll
        for (int t = 0; t < 16; ++t) {
            int k = 4 * t;
            s += x[t].x * W1[(k + 0) * 32 + n] + x[t].y * W1[(k + 1) * 32 + n]
               + x[t].z * W1[(k + 2) * 32 + n] + x[t].w * W1[(k + 3) * 32 + n];
        }
        float g = gelu_exact(s);
        o0 += g * W2[n * 3 + 0];
        o1 += g * W2[n * 3 + 1];
        o2 += g * W2[n * 3 + 2];
    }
    out[(size_t)j * 3 + 0] = o0;
    out[(size_t)j * 3 + 1] = o1;
    out[(size_t)j * 3 + 2] = o2;
}

extern "C" void kernel_launch(void* const* d_in, const int* in_sizes, int n_in,
                              void* d_out, int out_size, void* d_ws, size_t ws_size,
                              hipStream_t stream) {
    const float* seg     = (const float*)d_in[0];
    const float* coord   = (const float*)d_in[1];
    const int*   order   = (const int*)d_in[2];
    const int*   inverse = (const int*)d_in[3];
    const int*   gcoord  = (const int*)d_in[4];
    const float* W_in    = (const float*)d_in[5];
    const float* b_in    = (const float*)d_in[6];
    const float* g1      = (const float*)d_in[7];
    const float* be1     = (const float*)d_in[8];
    const float* W_qkv   = (const float*)d_in[9];
    const float* b_qkv   = (const float*)d_in[10];
    const float* rpe     = (const float*)d_in[11];
    const float* W_attn  = (const float*)d_in[12];
    const float* b_attn  = (const float*)d_in[13];
    const float* g2      = (const float*)d_in[14];
    const float* be2     = (const float*)d_in[15];
    const float* W_f1    = (const float*)d_in[16];
    const float* b_f1    = (const float*)d_in[17];
    const float* W_f2    = (const float*)d_in[18];
    const float* b_f2    = (const float*)d_in[19];
    const float* W_h1    = (const float*)d_in[20];
    const float* b_h1    = (const float*)d_in[21];
    const float* W_h2    = (const float*)d_in[22];
    const float* b_h2    = (const float*)d_in[23];

    char* ws = (char*)d_ws;
    float*    feat  = (float*)(ws + 0);                       // N*64*4
    float*    feat2 = (float*)(ws + 49152000);                // N*64*4
    _Float16* qbuf  = (_Float16*)(ws + 98304000);             // N*64*2
    _Float16* kbuf  = (_Float16*)(ws + 122880000);
    _Float16* vbuf  = (_Float16*)(ws + 147456000);
    _Float16* obuf  = (_Float16*)(ws + 172032000);
    _Float16* wbuf  = (_Float16*)(ws + 196608000);            // 49152 halves
    _Float16* qkvT  = wbuf;
    _Float16* attnT = wbuf + 12288;
    _Float16* f1T   = wbuf + 16384;
    _Float16* f2T   = wbuf + 32768;
    _Float16* hbuf  = obuf;      // reuse: obuf consumed by k_attnproj before k_ln2 writes
    float*    featF = feat;      // reuse: feat consumed by k_attnproj before k_ffn writes

    k_wprep   <<<192,  256, 0, stream>>>(W_qkv, W_attn, W_f1, W_f2, wbuf);
    k_inproj  <<<750,  256, 0, stream>>>(seg, coord, W_in, b_in, feat);
    k_qkv     <<<1500, 256, 0, stream>>>(feat, order, g1, be1, qkvT, b_qkv, qbuf, kbuf, vbuf);
    k_attn    <<<4000, 128, 0, stream>>>(order, gcoord, rpe, qbuf, kbuf, vbuf, obuf);
    k_attnproj<<<1500, 256, 0, stream>>>(obuf, inverse, feat, attnT, b_attn, feat2);
    k_ln2     <<<750,  256, 0, stream>>>(feat2, g2, be2, hbuf);
    k_ffn     <<<3000, 128, 0, stream>>>(hbuf, f1T, b_f1, f2T, b_f2, feat2, featF);
    k_head    <<<750,  256, 0, stream>>>(featF, W_h1, b_h1, W_h2, b_h2, (float*)d_out);
}
